// GPTLanguageModel_33457795236111
// MI455X (gfx1250) — matrix operand order repaired and checked by reading, not compiled, not run on hardware
//
#include <hip/hip_runtime.h>

// ---------------------------------------------------------------------------
// GPT forward for MI455X (gfx1250, wave32, WMMA 16x16x32 bf16, f32 accum)
// V=50257 D=384 H=6 HS=64 L=6 T=256 B=8   (M = B*T = 2048 rows)
// ---------------------------------------------------------------------------

typedef __bf16 bf16x16 __attribute__((ext_vector_type(16)));
typedef float  f32x8   __attribute__((ext_vector_type(8)));

union Frag {                 // one WMMA A or B operand (8 VGPRs)
    bf16x16        v;
    uint4          q[2];
    unsigned int   u[8];
    unsigned short s[16];
};

__device__ inline unsigned short f2bf(float f) {   // RNE f32 -> bf16
    union { float f; unsigned u; } x; x.f = f;
    unsigned r = x.u + 0x7fffu + ((x.u >> 16) & 1u);
    return (unsigned short)(r >> 16);
}

__device__ inline f32x8 zero8() {
    f32x8 z;
#pragma unroll
    for (int i = 0; i < 8; ++i) z[i] = 0.0f;
    return z;
}

// ---------------------------------------------------------------------------
// Repack an f32 weight matrix (K x N, element offset nbase(n) + k*strideK,
// nbase = nblock ? (n/64)*nblock + (n%64) : n) into bf16 WMMA-B fragment
// order: Bp[((tn*(K/32)+ks)*32 + lane)*16 + 0..15].  Tail columns zero-filled.
// ---------------------------------------------------------------------------
__global__ void repackB_k(const float* __restrict__ B, unsigned short* __restrict__ Bp,
                          int N, int K, int strideK, int nblock, int total)
{
    int tid = blockIdx.x * blockDim.x + threadIdx.x;
    if (tid >= total) return;
    const int lane = tid & 31;
    const int rest = tid >> 5;
    const int ksn  = K >> 5;
    const int ks   = rest % ksn;
    const int tn   = rest / ksn;
    const int n    = tn * 16 + (lane & 15);
    const int koffB = (lane >> 4) * 16;
    const bool ok  = (n < N);
    int nbase = nblock ? (n >> 6) * nblock + (n & 63) : n;

    Frag o;
    int off = nbase + (ks * 32 + koffB) * strideK;
#pragma unroll
    for (int v = 0; v < 8; ++v) {
        float f0 = ok ? B[off] : 0.0f;
        float f1 = ok ? B[off + strideK] : 0.0f;
        o.s[2 * v]     = f2bf(f0);
        o.s[2 * v + 1] = f2bf(f1);
        off += 2 * strideK;
    }
    uint4* dst = (uint4*)(Bp + (size_t)tid * 16);
    dst[0] = o.q[0];
    dst[1] = o.q[1];
}

// ---------------------------------------------------------------------------
// Packed GEMM, 32x32 register blocking: each wave computes a 32x32 block
// (2 M-subtiles x 2 N-subtiles).  Per K-step: 2 A frags + 2 B frags
// (8x b128 loads) feed 4 v_wmma_f32_16x16x32_bf16 -> 2 loads/WMMA.
// Requires M % 32 == 0, N % 32 == 0 (true for all layer GEMMs).
// ---------------------------------------------------------------------------
__global__ __launch_bounds__(32)
void gemm_pk_k(const unsigned short* __restrict__ A,
               const unsigned short* __restrict__ Bp,
               const float* __restrict__ bias,
               const float* __restrict__ resid,
               float* __restrict__ outF,
               unsigned short* __restrict__ outB,
               int M, int N, int K, int relu)
{
    const int lane  = threadIdx.x;
    const int tN    = blockIdx.x;          // 32-col block
    const int tM    = blockIdx.y;          // 32-row block
    const int mloc  = lane & 15;
    const int hi    = lane >> 4;
    const int koffA = hi * 16;
    const int ksn   = K >> 5;

    const unsigned short* ap0 = A + (size_t)(tM * 32 + mloc) * K + koffA;
    const unsigned short* ap1 = ap0 + (size_t)16 * K;
    const unsigned short* bp0 = Bp + ((size_t)(2 * tN) * ksn * 32 + lane) * 16;
    const unsigned short* bp1 = bp0 + (size_t)ksn * 32 * 16;

    f32x8 acc00 = zero8(), acc01 = zero8(), acc10 = zero8(), acc11 = zero8();

    for (int ks = 0; ks < ksn; ++ks) {
        Frag a0, a1, b0, b1;
        a0.q[0] = *(const uint4*)(ap0);
        a0.q[1] = *(const uint4*)(ap0 + 8);
        a1.q[0] = *(const uint4*)(ap1);
        a1.q[1] = *(const uint4*)(ap1 + 8);
        b0.q[0] = *(const uint4*)(bp0);
        b0.q[1] = *(const uint4*)(bp0 + 8);
        b1.q[0] = *(const uint4*)(bp1);
        b1.q[1] = *(const uint4*)(bp1 + 8);
        ap0 += 32; ap1 += 32; bp0 += 32 * 16; bp1 += 32 * 16;
        acc00 = __builtin_amdgcn_wmma_f32_16x16x32_bf16(false, a0.v, false, b0.v,
                                                        (short)0, acc00, false, false);
        acc01 = __builtin_amdgcn_wmma_f32_16x16x32_bf16(false, a0.v, false, b1.v,
                                                        (short)0, acc01, false, false);
        acc10 = __builtin_amdgcn_wmma_f32_16x16x32_bf16(false, a1.v, false, b0.v,
                                                        (short)0, acc10, false, false);
        acc11 = __builtin_amdgcn_wmma_f32_16x16x32_bf16(false, a1.v, false, b1.v,
                                                        (short)0, acc11, false, false);
    }

    const int n0 = tN * 32 + mloc;
    const int n1 = n0 + 16;
    const float bv0 = bias ? bias[n0] : 0.0f;
    const float bv1 = bias ? bias[n1] : 0.0f;

#pragma unroll
    for (int s = 0; s < 2; ++s) {
        const f32x8& cl = s ? acc10 : acc00;    // (row-sub s, col-sub 0)
        const f32x8& cr = s ? acc11 : acc01;    // (row-sub s, col-sub 1)
#pragma unroll
        for (int r = 0; r < 8; ++r) {
            int row = tM * 32 + s * 16 + r + 8 * hi;
            size_t o0 = (size_t)row * N + n0;
            size_t o1 = (size_t)row * N + n1;
            float v0 = cl[r] + bv0;
            float v1 = cr[r] + bv1;
            if (relu)  { v0 = fmaxf(v0, 0.0f); v1 = fmaxf(v1, 0.0f); }
            if (resid) { v0 += resid[o0];      v1 += resid[o1]; }
            if (outF)  { outF[o0] = v0;        outF[o1] = v1; }
            if (outB)  { outB[o0] = f2bf(v0);  outB[o1] = f2bf(v1); }
        }
    }
}

// ---------------------------------------------------------------------------
// LM-head GEMM: A(bf16) x B(f32 row-major K x N) + bias, ragged N.
// Address clamp (no predication): B lane n only feeds output column n,
// and out-of-range columns are never stored.  Output-write bound.
// ---------------------------------------------------------------------------
__global__ __launch_bounds__(32)
void gemm_lm_k(const unsigned short* __restrict__ A,
               const float* __restrict__ B,
               const float* __restrict__ bias,
               float* __restrict__ outF,
               int M, int N, int K)
{
    const int lane  = threadIdx.x;
    const int tN    = blockIdx.x;
    const int tM    = blockIdx.y;
    const int mloc  = lane & 15;
    const int hi    = lane >> 4;
    const int koffA = hi * 16;
    const int koffB = hi * 16;
    const int m     = tM * 16 + mloc;
    const int n     = tN * 16 + mloc;
    const bool nOK  = (n < N);
    const int  nc   = nOK ? n : (N - 1);           // clamped load column

    const unsigned short* ap = A + (size_t)m * K + koffA;

    f32x8 acc = zero8();
    int ro = koffB * N;
    for (int k0 = 0; k0 < K; k0 += 32) {
        Frag a, b;
        a.q[0] = *(const uint4*)(ap);
        a.q[1] = *(const uint4*)(ap + 8);
        ap += 32;
        int o = nc + ro;
#pragma unroll
        for (int v = 0; v < 8; ++v) {              // coalesced across lanes
            float f0 = B[o];
            float f1 = B[o + N];
            b.s[2 * v]     = f2bf(f0);
            b.s[2 * v + 1] = f2bf(f1);
            o += 2 * N;
        }
        ro += 32 * N;
        acc = __builtin_amdgcn_wmma_f32_16x16x32_bf16(false, a.v, false, b.v,
                                                      (short)0, acc, false, false);
    }

    if (nOK) {
        const float bv = bias ? bias[n] : 0.0f;
#pragma unroll
        for (int r = 0; r < 8; ++r) {
            int row = tM * 16 + r + 8 * hi;
            outF[(size_t)row * N + n] = acc[r] + bv;
        }
    }
}

// ---------------------------------------------------------------------------
// LayerNorm: one wave per row of 384; shuffle reductions (wave32).
// ---------------------------------------------------------------------------
__global__ __launch_bounds__(32)
void layernorm_k(const float* __restrict__ x, const float* __restrict__ g,
                 const float* __restrict__ bta, unsigned short* __restrict__ out)
{
    const int row = blockIdx.x, lane = threadIdx.x;
    float v[12], s = 0.0f;
#pragma unroll
    for (int j = 0; j < 12; ++j) { v[j] = x[(size_t)row * 384 + lane + 32 * j]; s += v[j]; }
#pragma unroll
    for (int m = 16; m >= 1; m >>= 1) s += __shfl_xor(s, m, 32);
    const float mu = s * (1.0f / 384.0f);
    float q = 0.0f;
#pragma unroll
    for (int j = 0; j < 12; ++j) { float d = v[j] - mu; q += d * d; }
#pragma unroll
    for (int m = 16; m >= 1; m >>= 1) q += __shfl_xor(q, m, 32);
    const float inv = rsqrtf(q * (1.0f / 384.0f) + 1e-5f);
#pragma unroll
    for (int j = 0; j < 12; ++j) {
        int col = lane + 32 * j;
        out[(size_t)row * 384 + col] = f2bf((v[j] - mu) * inv * g[col] + bta[col]);
    }
}

// ---------------------------------------------------------------------------
// Embedding gather: x[i,d] = tok_emb[idx[i], d]
// ---------------------------------------------------------------------------
__global__ void embed_k(const int* __restrict__ idx, const float* __restrict__ emb,
                        float* __restrict__ x, int total)
{
    int tid = blockIdx.x * blockDim.x + threadIdx.x;
    if (tid >= total) return;
    int row = tid / 384, d = tid - row * 384;
    x[tid] = emb[(size_t)idx[row] * 384 + d];
}

// ---------------------------------------------------------------------------
// RoPE exactly as the reference: out[2j] = -q[2j+1]*cosA[j], out[2j+1]=q[2j]*sinA[j]
// sinA[t][j] = emb[t][j], cosA[t][j] = emb[t][32+j] (emb interleaved sin/cos).
// ---------------------------------------------------------------------------
__global__ void rope_k(const float* __restrict__ qf, const float* __restrict__ kf,
                       unsigned short* __restrict__ qb, unsigned short* __restrict__ kb)
{
    int tid = blockIdx.x * blockDim.x + threadIdx.x;
    if (tid >= 2048 * 6 * 32) return;
    int row = tid / 192, rem = tid - row * 192;
    int h = rem >> 5, j = rem & 31;
    float tf = (float)(row & 255);                 // t = row % T
    const float c = -9.210340371976184f / 64.0f;   // -ln(10000)/HS
    int i = j >> 1;
    float a1 = tf * __expf((float)(2 * i) * c);
    float a2 = tf * __expf((float)(2 * (16 + i)) * c);
    float sA = (j & 1) ? __cosf(a1) : __sinf(a1);
    float cA = (j & 1) ? __cosf(a2) : __sinf(a2);
    size_t base = (size_t)row * 384 + h * 64 + 2 * j;
    float qe = qf[base], qo = qf[base + 1];
    qb[base] = f2bf(-qo * cA); qb[base + 1] = f2bf(qe * sA);
    float ke = kf[base], ko = kf[base + 1];
    kb[base] = f2bf(-ko * cA); kb[base + 1] = f2bf(ke * sA);
}

// ---------------------------------------------------------------------------
// Attention: one wave per (rowtile, head, batch).  S = QK^T (WMMA), causal
// mask + scale into LDS, wave softmax, P (bf16 in LDS) @ V (WMMA) -> att bf16.
// LDS: 16x256 f32 scores + 16x256 bf16 probs = 24 KB.
// ---------------------------------------------------------------------------
__global__ __launch_bounds__(32)
void attn_k(const unsigned short* __restrict__ qb, const unsigned short* __restrict__ kb,
            const unsigned short* __restrict__ vb, unsigned short* __restrict__ attb)
{
    __shared__ float          sS[16 * 256];
    __shared__ unsigned short sP[16 * 256];
    const int lane = threadIdx.x;
    const int rt = blockIdx.x, h = blockIdx.y, b = blockIdx.z;
    const int mloc = lane & 15, hi = lane >> 4;
    const int koffA = hi * 16, koffB = hi * 16;
    const float scale = 0.125f;                    // 64^-0.5
    const size_t bt0 = (size_t)b * 256;

    // Q row-tile as two A fragments (K = 64 -> 2 steps of 32)
    Frag aq[2];
    const unsigned short* qp = qb + (bt0 + rt * 16 + mloc) * 384 + h * 64 + koffA;
#pragma unroll
    for (int ks = 0; ks < 2; ++ks) {
        aq[ks].q[0] = *(const uint4*)(qp + ks * 32);
        aq[ks].q[1] = *(const uint4*)(qp + ks * 32 + 8);
    }

    // scores
    for (int kt = 0; kt < 16; ++kt) {
        f32x8 acc = zero8();
        const unsigned short* kp = kb + (bt0 + kt * 16 + mloc) * 384 + h * 64 + koffB;
#pragma unroll
        for (int ks = 0; ks < 2; ++ks) {
            Frag bk;                               // K^T: hs pairs contiguous -> b128
            bk.q[0] = *(const uint4*)(kp + ks * 32);
            bk.q[1] = *(const uint4*)(kp + ks * 32 + 8);
            acc = __builtin_amdgcn_wmma_f32_16x16x32_bf16(false, aq[ks].v, false, bk.v,
                                                          (short)0, acc, false, false);
        }
#pragma unroll
        for (int r = 0; r < 8; ++r) {
            int rl = r + 8 * hi;
            int tg = rt * 16 + rl, sg = kt * 16 + mloc;
            float val = acc[r] * scale;
            if (sg > tg) val = -1.0e30f;           // causal mask
            sS[rl * 256 + sg] = val;
        }
    }
    __syncthreads();

    // softmax over each of the 16 rows (256 cols, 8 per lane)
    for (int r = 0; r < 16; ++r) {
        float e[8], mx = -1.0e30f;
#pragma unroll
        for (int j = 0; j < 8; ++j) { e[j] = sS[r * 256 + lane + 32 * j]; mx = fmaxf(mx, e[j]); }
#pragma unroll
        for (int m = 16; m >= 1; m >>= 1) mx = fmaxf(mx, __shfl_xor(mx, m, 32));
        float sm = 0.0f;
#pragma unroll
        for (int j = 0; j < 8; ++j) { e[j] = __expf(e[j] - mx); sm += e[j]; }
#pragma unroll
        for (int m = 16; m >= 1; m >>= 1) sm += __shfl_xor(sm, m, 32);
        float inv = 1.0f / sm;
#pragma unroll
        for (int j = 0; j < 8; ++j) sP[r * 256 + lane + 32 * j] = f2bf(e[j] * inv);
    }
    __syncthreads();

    // att = P(16x256) @ V(256x64): preload P A-fragments from LDS (ds_load_b128)
    Frag ap[8];
#pragma unroll
    for (int ks = 0; ks < 8; ++ks) {
        const unsigned short* pp = sP + mloc * 256 + ks * 32 + koffA;
        ap[ks].q[0] = *(const uint4*)(pp);
        ap[ks].q[1] = *(const uint4*)(pp + 8);
    }
#pragma unroll
    for (int nt = 0; nt < 4; ++nt) {
        f32x8 acc = zero8();
        for (int ks = 0; ks < 8; ++ks) {
            Frag bv_;
#pragma unroll
            for (int v = 0; v < 8; ++v) {          // V: K dim = key pos, stride 384
                int kk = ks * 32 + koffB + 2 * v;
                size_t r0 = (bt0 + kk) * 384 + h * 64 + nt * 16 + mloc;
                bv_.s[2 * v]     = vb[r0];
                bv_.s[2 * v + 1] = vb[r0 + 384];
            }
            acc = __builtin_amdgcn_wmma_f32_16x16x32_bf16(false, ap[ks].v, false, bv_.v,
                                                          (short)0, acc, false, false);
        }
#pragma unroll
        for (int r = 0; r < 8; ++r) {
            int t = rt * 16 + r + 8 * hi;
            attb[(bt0 + t) * 384 + h * 64 + nt * 16 + mloc] = f2bf(acc[r]);
        }
    }
}

// ---------------------------------------------------------------------------
// host-side launch
// ---------------------------------------------------------------------------
static inline char* wsalign(char*& p, size_t bytes) {
    char* r = p;
    p += (bytes + 255) & ~(size_t)255;
    return r;
}

extern "C" void kernel_launch(void* const* d_in, const int* in_sizes, int n_in,
                              void* d_out, int out_size, void* d_ws, size_t ws_size,
                              hipStream_t stream)
{
    const int*   idx     = (const int*)  d_in[0];
    const float* tok_emb = (const float*)d_in[1];
    const float* Wq      = (const float*)d_in[2];
    const float* Wk      = (const float*)d_in[3];
    const float* Wv      = (const float*)d_in[4];
    const float* Wp      = (const float*)d_in[5];
    const float* bp      = (const float*)d_in[6];
    const float* W1      = (const float*)d_in[7];
    const float* b1      = (const float*)d_in[8];
    const float* W2      = (const float*)d_in[9];
    const float* b2      = (const float*)d_in[10];
    const float* ln1_g   = (const float*)d_in[11];
    const float* ln1_b   = (const float*)d_in[12];
    const float* ln2_g   = (const float*)d_in[13];
    const float* ln2_b   = (const float*)d_in[14];
    const float* lnf_g   = (const float*)d_in[15];
    const float* lnf_b   = (const float*)d_in[16];
    const float* lmW     = (const float*)d_in[17];
    const float* lmb     = (const float*)d_in[18];
    float* out = (float*)d_out;

    const int M = 2048, D = 384, Dm = 1536, V = 50257;

    char* p = (char*)d_ws;
    float*          x   = (float*)         wsalign(p, (size_t)M * D * 4);
    unsigned short* xn  = (unsigned short*)wsalign(p, (size_t)M * D * 2);
    float*          qf  = (float*)         wsalign(p, (size_t)M * D * 4);
    float*          kf  = (float*)         wsalign(p, (size_t)M * D * 4);
    unsigned short* qb  = (unsigned short*)wsalign(p, (size_t)M * D * 2);
    unsigned short* kb  = (unsigned short*)wsalign(p, (size_t)M * D * 2);
    unsigned short* vb  = (unsigned short*)wsalign(p, (size_t)M * D * 2);
    unsigned short* ab  = (unsigned short*)wsalign(p, (size_t)M * D * 2);
    unsigned short* hb  = (unsigned short*)wsalign(p, (size_t)M * Dm * 2);
    // per-layer packed weights (reused each layer)
    unsigned short* pq  = (unsigned short*)wsalign(p, (size_t)D * D * 2);
    unsigned short* pk  = (unsigned short*)wsalign(p, (size_t)D * D * 2);
    unsigned short* pv  = (unsigned short*)wsalign(p, (size_t)D * D * 2);
    unsigned short* pp_ = (unsigned short*)wsalign(p, (size_t)D * D * 2);
    unsigned short* p1  = (unsigned short*)wsalign(p, (size_t)D * Dm * 2);
    unsigned short* p2  = (unsigned short*)wsalign(p, (size_t)Dm * D * 2);

    const dim3 wave(32);
    const dim3 gD(D / 32, M / 32);        // 12 x 64   (32x32 blocks)
    const dim3 gDm(Dm / 32, M / 32);      // 48 x 64
    const dim3 gV((V + 15) / 16, M / 16); // 3142 x 128 (16x16, ragged N)
    const dim3 gAttn(16, 6, 8);

    const int tpD  = (D / 16)  * (D >> 5)  * 32;  // repack threads: N=384,K=384
    const int tpD1 = (Dm / 16) * (D >> 5)  * 32;  // N=1536,K=384
    const int tpD2 = (D / 16)  * (Dm >> 5) * 32;  // N=384,K=1536

    embed_k<<<(M * D + 255) / 256, 256, 0, stream>>>(idx, tok_emb, x, M * D);

    for (int l = 0; l < 6; ++l) {
        const float* wq = Wq + (size_t)l * 6 * 384 * 64;
        const float* wk = Wk + (size_t)l * 6 * 384 * 64;
        const float* wv = Wv + (size_t)l * 6 * 384 * 64;
        const float* wp = Wp + (size_t)l * 384 * 384;
        const float* w1 = W1 + (size_t)l * 384 * 1536;
        const float* w2 = W2 + (size_t)l * 1536 * 384;

        // pack this layer's weights into fragment order (head-blocked for QKV)
        repackB_k<<<(tpD  + 255) / 256, 256, 0, stream>>>(wq, pq,  D,  D,  64, 24576, tpD);
        repackB_k<<<(tpD  + 255) / 256, 256, 0, stream>>>(wk, pk,  D,  D,  64, 24576, tpD);
        repackB_k<<<(tpD  + 255) / 256, 256, 0, stream>>>(wv, pv,  D,  D,  64, 24576, tpD);
        repackB_k<<<(tpD  + 255) / 256, 256, 0, stream>>>(wp, pp_, D,  D,  D,  0,     tpD);
        repackB_k<<<(tpD1 + 255) / 256, 256, 0, stream>>>(w1, p1,  Dm, D,  Dm, 0,     tpD1);
        repackB_k<<<(tpD2 + 255) / 256, 256, 0, stream>>>(w2, p2,  D,  Dm, D,  0,     tpD2);

        layernorm_k<<<M, wave, 0, stream>>>(x, ln1_g + l * D, ln1_b + l * D, xn);

        gemm_pk_k<<<gD, wave, 0, stream>>>(xn, pq, nullptr, nullptr, qf, nullptr, M, D, D, 0);
        gemm_pk_k<<<gD, wave, 0, stream>>>(xn, pk, nullptr, nullptr, kf, nullptr, M, D, D, 0);
        gemm_pk_k<<<gD, wave, 0, stream>>>(xn, pv, nullptr, nullptr, nullptr, vb, M, D, D, 0);

        rope_k<<<(M * 6 * 32 + 255) / 256, 256, 0, stream>>>(qf, kf, qb, kb);

        attn_k<<<gAttn, wave, 0, stream>>>(qb, kb, vb, ab);

        // x += att @ Wp + bp   (in-place residual)
        gemm_pk_k<<<gD, wave, 0, stream>>>(ab, pp_, bp + l * D, x, x, nullptr, M, D, D, 0);

        layernorm_k<<<M, wave, 0, stream>>>(x, ln2_g + l * D, ln2_b + l * D, xn);

        // h = relu(xn @ W1 + b1) -> bf16
        gemm_pk_k<<<gDm, wave, 0, stream>>>(xn, p1, b1 + l * Dm, nullptr, nullptr, hb, M, Dm, D, 1);
        // x += h @ W2 + b2
        gemm_pk_k<<<gD, wave, 0, stream>>>(hb, p2, b2 + l * D, x, x, nullptr, M, D, Dm, 0);
    }

    layernorm_k<<<M, wave, 0, stream>>>(x, lnf_g, lnf_b, xn);

    // logits = xn @ lmW + lmb  (ragged N, clamped loads, coalesced f32 B)
    gemm_lm_k<<<gV, wave, 0, stream>>>(xn, lmW, lmb, out, M, V, D);
}
